// StackLSTMBatch_58282706207126
// MI455X (gfx1250) — compile-verified, matrix operand order checked by reading, not run on hardware
//
#include <hip/hip_runtime.h>
#include <hip/hip_bf16.h>
#include <math.h>

// ---------------- problem constants ----------------
#define Tn   512
#define Bn   256
#define INn  128
#define Hn   256
#define Sp1  129          // stack slots (S+1)
#define Kdim 640          // IN + H + H (concatenated GEMM K)
#define ZSTR 648          // LDS row stride (pad: 324 dwords % 64 banks = 4 -> conflict free)
#define NKT  20           // Kdim / 32
#define NCT  64           // 4*H / 16 column tiles
#define BH   (Bn*Hn)

typedef __attribute__((ext_vector_type(16))) __bf16 v16bf;
typedef __attribute__((ext_vector_type(8)))  __bf16 v8bf;
typedef __attribute__((ext_vector_type(8)))  float  v8f;

// =====================================================================
// Pack Wbig[640][1024] (bf16) directly into per-lane WMMA B-fragments:
// frag f = kt*64 + ct holds 32 lanes x 16 bf16, contiguous (32B per lane).
// Column c = ct*16 + (lane&15); gate = c/256, h = c%256.
// K element index mirrors the documented 16-bit A layout:
//   k = kt*32 + (lane<16 ? 0 : 8) + (j<8 ? j : j+8)
// =====================================================================
__global__ __launch_bounds__(256) void stklstm_pack_w(
    const float* __restrict__ w_xi, const float* __restrict__ w_hi, const float* __restrict__ w_ci,
    const float* __restrict__ w_xf, const float* __restrict__ w_hf, const float* __restrict__ w_cf,
    const float* __restrict__ w_xc, const float* __restrict__ w_xo, const float* __restrict__ w_ho,
    const float* __restrict__ w_co, __bf16* __restrict__ Bpk)
{
    int idx = blockIdx.x * 256 + threadIdx.x;          // NKT*NCT*32*16 = 655360 elements
    if (idx >= NKT * NCT * 32 * 16) return;
    int j    = idx & 15;
    int lane = (idx >> 4) & 31;
    int f    = idx >> 9;
    int kt   = f >> 6;
    int ct   = f & 63;
    int col  = ct * 16 + (lane & 15);
    int k    = kt * 32 + ((lane < 16) ? 0 : 8) + ((j < 8) ? j : j + 8);
    int gate = col >> 8;                                // 0..3
    int h    = col & 255;
    float v = 0.0f;
    if (k < INn) {
        const float* wx = (gate == 0) ? w_xi : (gate == 1) ? w_xf : (gate == 2) ? w_xc : w_xo;
        v = wx[h * INn + k];
    } else if (k < INn + Hn) {
        // NOTE: gate 2 (cell candidate) reuses w_hf, faithful to reference
        const float* wh = (gate == 0) ? w_hi : (gate == 3) ? w_ho : w_hf;
        v = wh[h * Hn + (k - INn)];
    } else {
        if (gate != 2) {
            const float* wc = (gate == 0) ? w_ci : (gate == 1) ? w_cf : w_co;
            v = wc[h * Hn + (k - INn - Hn)];
        }
    }
    Bpk[(size_t)f * 512 + lane * 16 + j] = (__bf16)v;
}

// =====================================================================
// Stack-pointer prefix sum + push mask (T x B, tiny).
// =====================================================================
__global__ __launch_bounds__(64) void stklstm_scan_ops(
    const int* __restrict__ ops, int* __restrict__ pts, int* __restrict__ opm)
{
    int b = blockIdx.x * 64 + threadIdx.x;
    if (b >= Bn) return;
    int s = 0;
    for (int t = 0; t < Tn; ++t) {
        s += ops[t * Bn + b];                           // pts[t] = cumsum through ops[t]
        pts[t * Bn + b] = s;
        opm[t * Bn + b] = (ops[(t + 1) * Bn + b] > 0) ? 1 : 0;
    }
}

// =====================================================================
// Zero both stacks, seed slot 0 with h0/c0 broadcast (reference zero-init
// matters: prev_h can read slot (pt-1) mod 129 before it is ever written).
// =====================================================================
__global__ __launch_bounds__(256) void stklstm_init_stacks(
    float* __restrict__ h_stack, float* __restrict__ c_stack,
    const float* __restrict__ h0, const float* __restrict__ c0)
{
    size_t idx = (size_t)blockIdx.x * 256 + threadIdx.x;
    const size_t SZ = (size_t)Sp1 * BH;
    if (idx >= SZ) return;
    size_t slot = idx / (size_t)BH;
    int h = (int)(idx & (Hn - 1));
    h_stack[idx] = (slot == 0) ? h0[h] : 0.0f;
    c_stack[idx] = (slot == 0) ? c0[h] : 0.0f;
}

// =====================================================================
// One recurrent step. Grid = 32 WGs x 256 threads (8 wave32s).
// WG (blk) owns batch rows m0..m0+15 (m0 = (blk>>1)*16).
// Wave w (global 0..255): mt = w>>4 (== blk>>1), nt = w&15.
// Each wave: 4 accumulators = one 16x16 (b,h) tile per gate.
// K loop: 20 x 4 = 80 v_wmma_f32_16x16x32_bf16 per wave, with the 4
// B-fragments for kt+1 double-buffered (issued before consuming kt's)
// so each global load has ~4 WMMAs of slack -- critical at only
// 2 waves/SIMD of occupancy.
// =====================================================================
__global__ __launch_bounds__(256) void stklstm_step(
    const float* __restrict__ x_t,                      // [B, IN] (inputs + t*B*IN)
    const float* __restrict__ b_i, const float* __restrict__ b_f,
    const float* __restrict__ b_c, const float* __restrict__ b_o,
    const __bf16* __restrict__ Bpk,
    float* __restrict__ h_stack, float* __restrict__ c_stack,
    const int* __restrict__ pts_t, const int* __restrict__ opm_t,
    float* __restrict__ out_t)                          // [B, H] (out + t*B*H)
{
    __shared__ __bf16 zS[16 * ZSTR];
    __shared__ int    ptS[16];

    const int tid = threadIdx.x;
    const int m0  = (blockIdx.x >> 1) * 16;

    if (tid < 16) ptS[tid] = pts_t[m0 + tid];
    __syncthreads();

    // ---- gather Z = [x | cur_h | cur_c] rows m0..m0+15, f32 -> bf16 into LDS
    for (int idx = tid; idx < 16 * Kdim; idx += 256) {
        int row = idx / Kdim;
        int k   = idx - row * Kdim;
        int b   = m0 + row;
        float v;
        if (k < INn) {
            v = x_t[b * INn + k];
        } else if (k < INn + Hn) {
            v = h_stack[(size_t)ptS[row] * BH + b * Hn + (k - INn)];
        } else {
            v = c_stack[(size_t)ptS[row] * BH + b * Hn + (k - INn - Hn)];
        }
        zS[row * ZSTR + k] = (__bf16)v;
    }
    __syncthreads();

    // ---- GEMM: G[16,1024] tile per wave (16 rows x 16 cols x 4 gates)
    const int wave  = tid >> 5;
    const int lane  = tid & 31;
    const int wglob = blockIdx.x * 8 + wave;
    const int nt    = wglob & 15;
    const int arow  = lane & 15;
    const int khalf = (lane < 16) ? 0 : 8;              // A layout: lanes 16-31 hold K+8 half

    v8f acc0 = {}, acc1 = {}, acc2 = {}, acc3 = {};

    // per-lane base into the B-fragment stream for this wave's column tile
    const __bf16* bbase = Bpk + (size_t)lane * 16 + (size_t)nt * 512;

    // software pipeline: bw holds kt's fragments, bn holds kt+1's
    v16bf bw0, bw1, bw2, bw3;
    {
        const __bf16* bp = bbase;                       // kt = 0
        bw0 = *(const v16bf*)(bp + (size_t)(0 * 16) * 512);
        bw1 = *(const v16bf*)(bp + (size_t)(1 * 16) * 512);
        bw2 = *(const v16bf*)(bp + (size_t)(2 * 16) * 512);
        bw3 = *(const v16bf*)(bp + (size_t)(3 * 16) * 512);
    }

    #pragma unroll
    for (int kt = 0; kt < NKT; ++kt) {
        // 1) issue next K-slice's B loads first (long-latency L2 hits)
        v16bf bn0, bn1, bn2, bn3;
        if (kt + 1 < NKT) {
            const __bf16* bp = bbase + (size_t)(kt + 1) * NCT * 512;
            bn0 = *(const v16bf*)(bp + (size_t)(0 * 16) * 512);
            bn1 = *(const v16bf*)(bp + (size_t)(1 * 16) * 512);
            bn2 = *(const v16bf*)(bp + (size_t)(2 * 16) * 512);
            bn3 = *(const v16bf*)(bp + (size_t)(3 * 16) * 512);
            if (kt + 2 < NKT)                           // global_prefetch_b8 for kt+2
                __builtin_prefetch(bp + (size_t)NCT * 512, 0, 1);
        }

        // 2) A fragment from LDS (two aligned 16B chunks: K {0..7},{16..23} of this half)
        const __bf16* ap = &zS[arow * ZSTR + kt * 32 + khalf];
        v8bf alo = *(const v8bf*)ap;
        v8bf ahi = *(const v8bf*)(ap + 16);
        v16bf a;
        #pragma unroll
        for (int i = 0; i < 8; ++i) { a[i] = alo[i]; a[i + 8] = ahi[i]; }

        // 3) consume current fragments (resident since previous iteration)
        acc0 = __builtin_amdgcn_wmma_f32_16x16x32_bf16(false, a, false, bw0, (short)0, acc0, false, false);
        acc1 = __builtin_amdgcn_wmma_f32_16x16x32_bf16(false, a, false, bw1, (short)0, acc1, false, false);
        acc2 = __builtin_amdgcn_wmma_f32_16x16x32_bf16(false, a, false, bw2, (short)0, acc2, false, false);
        acc3 = __builtin_amdgcn_wmma_f32_16x16x32_bf16(false, a, false, bw3, (short)0, acc3, false, false);

        if (kt + 1 < NKT) { bw0 = bn0; bw1 = bn1; bw2 = bn2; bw3 = bn3; }
    }

    // ---- elementwise gates + stack scatter + masked output select
    // C/D layout: lane 0-15 -> N=lane, VGPR r -> M=r; lane 16-31 -> N=lane-16, M=r+8
    const int h = nt * 16 + (lane & 15);
    const float bi = b_i[h], bf = b_f[h], bc = b_c[h], bo = b_o[h];

    #pragma unroll
    for (int r = 0; r < 8; ++r) {
        int mrow = (lane < 16) ? r : r + 8;
        int b    = m0 + mrow;
        int p    = ptS[mrow];
        size_t base = (size_t)b * Hn + h;

        float cc = c_stack[(size_t)p * BH + base];       // cur_c (slot p: read-only this step)
        float gi = acc0[r] + bi;
        float gf = acc1[r] + bf;
        float gc = acc2[r] + bc;
        float go = acc3[r] + bo;
        float ig = 1.0f / (1.0f + __expf(-gi));
        float fg = 1.0f / (1.0f + __expf(-gf));
        float og = 1.0f / (1.0f + __expf(-go));
        float cg = fg * cc + ig * tanhf(gc);
        float nh = og * tanhf(cg);

        h_stack[(size_t)(p + 1) * BH + base] = nh;       // scatter to slot p+1
        c_stack[(size_t)(p + 1) * BH + base] = cg;

        float ph = h_stack[(size_t)((p + Sp1 - 1) % Sp1) * BH + base];  // prev_h (slot untouched)
        out_t[base] = opm_t[b] ? nh : ph;
    }
}

// =====================================================================
// Host-side orchestration: pack + scan + init, then 512 step launches.
// All on `stream`; graph-capture safe (no malloc/sync).
// =====================================================================
extern "C" void kernel_launch(void* const* d_in, const int* in_sizes, int n_in,
                              void* d_out, int out_size, void* d_ws, size_t ws_size,
                              hipStream_t stream) {
    const float* inputs = (const float*)d_in[0];
    const int*   ops    = (const int*)  d_in[1];
    const float* w_xi = (const float*)d_in[2];  const float* b_i = (const float*)d_in[3];
    const float* w_hi = (const float*)d_in[4];  const float* w_ci = (const float*)d_in[5];
    const float* w_xf = (const float*)d_in[6];  const float* b_f = (const float*)d_in[7];
    const float* w_hf = (const float*)d_in[8];  const float* w_cf = (const float*)d_in[9];
    const float* w_xc = (const float*)d_in[10]; const float* b_c = (const float*)d_in[11];
    const float* w_xo = (const float*)d_in[12]; const float* b_o = (const float*)d_in[13];
    const float* w_ho = (const float*)d_in[14]; const float* w_co = (const float*)d_in[15];
    const float* h0   = (const float*)d_in[16]; const float* c0  = (const float*)d_in[17];
    float* out = (float*)d_out;

    // workspace layout (bytes)
    const size_t BPK_BYTES = (size_t)NKT * NCT * 512 * sizeof(__bf16); // 1,310,720
    const size_t STK_BYTES = (size_t)Sp1 * BH * sizeof(float);         // 33,816,576
    const size_t TB_BYTES  = (size_t)Tn * Bn * sizeof(int);            // 524,288
    char* ws = (char*)d_ws;
    __bf16* Bpk    = (__bf16*)(ws);
    float* h_stack = (float*)(ws + BPK_BYTES);
    float* c_stack = (float*)(ws + BPK_BYTES + STK_BYTES);
    int*   pts     = (int*)  (ws + BPK_BYTES + 2 * STK_BYTES);
    int*   opm     = (int*)  (ws + BPK_BYTES + 2 * STK_BYTES + TB_BYTES);
    (void)in_sizes; (void)n_in; (void)out_size; (void)ws_size;

    stklstm_pack_w<<<(NKT * NCT * 32 * 16 + 255) / 256, 256, 0, stream>>>(
        w_xi, w_hi, w_ci, w_xf, w_hf, w_cf, w_xc, w_xo, w_ho, w_co, Bpk);
    stklstm_scan_ops<<<(Bn + 63) / 64, 64, 0, stream>>>(ops, pts, opm);
    {
        size_t SZ = (size_t)Sp1 * BH;
        stklstm_init_stacks<<<(unsigned)((SZ + 255) / 256), 256, 0, stream>>>(h_stack, c_stack, h0, c0);
    }
    for (int t = 0; t < Tn; ++t) {
        stklstm_step<<<32, 256, 0, stream>>>(
            inputs + (size_t)t * Bn * INn,
            b_i, b_f, b_c, b_o, Bpk,
            h_stack, c_stack,
            pts + (size_t)t * Bn, opm + (size_t)t * Bn,
            out + (size_t)t * BH);
    }
}
